// ODEVAE_76201309765846
// MI455X (gfx1250) — compile-verified, matrix operand order checked by reading, not run on hardware
//
#include <hip/hip_runtime.h>
#include <hip/hip_bf16.h>
#include <math.h>

// ---------------- problem dims ----------------
#define Bsz 512
#define Tsz 64
#define Dsz 32
#define Hsz 256
#define Lsz 64
#define BT  32        // batch rows per workgroup (tile)
#define BLOCK 256     // 8 wave32s
#define NWAVES 8

typedef __attribute__((ext_vector_type(16))) __bf16 v16bf;
typedef __attribute__((ext_vector_type(8)))  float  v8f;

// K-index map for 16-bit A/B fragments (ISA 7.12.2):
// lane group g = lane>>4; element e<8 -> K=e+8g ; e>=8 -> K=e+8+8g
__device__ __forceinline__ int kmap(int e, int g) { return e + (e & 8) + (g << 3); }

// A fragment from bf16 LDS: per lane this is two contiguous 16B runs -> 2x ds_load_b128
__device__ __forceinline__ v16bf load_a_bf16(const __bf16* src, int ld, int mi, int kb, int lane) {
  int m = mi + (lane & 15);
  int g = (lane >> 4) & 1;
  v16bf a;
#pragma unroll
  for (int e = 0; e < 16; ++e) a[e] = src[m * ld + kb + kmap(e, g)];
  return a;
}
// B[k,n] = W[n,k]  (reference computes X @ W^T, W row-major (N,K)) -> 2x global_load_b128
__device__ __forceinline__ v16bf load_b_w(const __bf16* W, int ld, int nb, int kb, int lane) {
  int n = nb + (lane & 15);
  int g = (lane >> 4) & 1;
  v16bf b;
#pragma unroll
  for (int e = 0; e < 16; ++e) b[e] = W[n * ld + kb + kmap(e, g)];
  return b;
}
__device__ __forceinline__ v8f bias_init(const float* bias, int nb, int lane) {
  float bv = bias[nb + (lane & 15)];
  v8f c;
#pragma unroll
  for (int r = 0; r < 8; ++r) c[r] = bv;
  return c;
}
__device__ __forceinline__ float act_apply(float v, int ACT) {
  if (ACT == 1) return tanhf(v);
  if (ACT == 2) return fmaxf(v, 0.0f);
  if (ACT == 3) return 1.0f / (1.0f + expf(-v));
  return v;
}
// C/D layout: VGPR r, lane group g: M = mi + 8*g + r, N = nb + lane%16
template <int ACT>
__device__ __forceinline__ void store_c(float* dst, int ld, int mi, int nb, int lane, v8f c) {
  int n = nb + (lane & 15);
  int g = (lane >> 4) & 1;
#pragma unroll
  for (int r = 0; r < 8; ++r)
    dst[(mi + g * 8 + r) * ld + n] = act_apply(c[r], ACT);
}
template <int ACT>
__device__ __forceinline__ void store_c_bf(__bf16* dst, int ld, int mi, int nb, int lane, v8f c) {
  int n = nb + (lane & 15);
  int g = (lane >> 4) & 1;
#pragma unroll
  for (int r = 0; r < 8; ++r)
    dst[(mi + g * 8 + r) * ld + n] = (__bf16)act_apply(c[r], ACT);
}

#define WMMA_BF16(a, b, c) \
  __builtin_amdgcn_wmma_f32_16x16x32_bf16(false, (a), false, (b), (short)0, (c), false, false)

// dst_f32(MxN) = act(src_bf16(MxK) @ W(N,K)^T + bias)
template <int ACT>
__device__ void wmma_layer_f(const __bf16* src, const __bf16* W, const float* bias,
                             float* dst, int M, int N, int K) {
  const int lane = threadIdx.x & 31;
  const int wave = threadIdx.x >> 5;
  const int tn = N >> 4;
  const int ntiles = (M >> 4) * tn;
  for (int t = wave; t < ntiles; t += NWAVES) {
    int mi = (t / tn) << 4;
    int ni = (t % tn) << 4;
    v8f c = bias_init(bias, ni, lane);
    for (int k = 0; k < K; k += 32) {
      v16bf a = load_a_bf16(src, K, mi, k, lane);
      v16bf b = load_b_w(W, K, ni, k, lane);
      c = WMMA_BF16(a, b, c);
    }
    store_c<ACT>(dst, N, mi, ni, lane, c);
  }
}
// dst_bf16(MxN) = act(src_bf16(MxK) @ W(N,K)^T + bias)
template <int ACT>
__device__ void wmma_layer_b(const __bf16* src, const __bf16* W, const float* bias,
                             __bf16* dst, int M, int N, int K) {
  const int lane = threadIdx.x & 31;
  const int wave = threadIdx.x >> 5;
  const int tn = N >> 4;
  const int ntiles = (M >> 4) * tn;
  for (int t = wave; t < ntiles; t += NWAVES) {
    int mi = (t / tn) << 4;
    int ni = (t % tn) << 4;
    v8f c = bias_init(bias, ni, lane);
    for (int k = 0; k < K; k += 32) {
      v16bf a = load_a_bf16(src, K, mi, k, lane);
      v16bf b = load_b_w(W, K, ni, k, lane);
      c = WMMA_BF16(a, b, c);
    }
    store_c_bf<ACT>(dst, N, mi, ni, lane, c);
  }
}

// ---------------- prep: fp32 -> bf16 weight arenas in d_ws ----------------
__global__ void k_cvt(const float* __restrict__ src, __bf16* __restrict__ dst, int n) {
  int i = blockIdx.x * blockDim.x + threadIdx.x;
  if (i < n) dst[i] = (__bf16)src[i];
}
__global__ void k_pad_wih(const float* __restrict__ W, __bf16* __restrict__ dst) {
  // (3H,33) -> (3H,64) zero-padded so the GRU input GEMM has K=64
  int i = blockIdx.x * blockDim.x + threadIdx.x;
  if (i >= 768 * 64) return;
  int n = i >> 6, k = i & 63;
  dst[i] = (__bf16)(k < (Dsz + 1) ? W[n * (Dsz + 1) + k] : 0.0f);
}

// ---------------- GRU backward scan (persistent per batch-tile) ----------------
__global__ void __launch_bounds__(BLOCK)
k_gru(const float* __restrict__ x, const float* __restrict__ tglob,
      const __bf16* __restrict__ Wihp, const float* __restrict__ b_ih,
      const __bf16* __restrict__ Whh, const float* __restrict__ b_hh,
      float* __restrict__ hT) {
  extern __shared__ char smem[];
  float* rbuf = (float*)smem;                 // 32x256 sigmoid(r)        128KB..
  float* zbuf = rbuf + BT * Hsz;              // 32x256 sigmoid(z)
  float* ginb = zbuf + BT * Hsz;              // 32x256 gi_n
  float* ghnb = ginb + BT * Hsz;              // 32x256 gh_n
  float* hbuf = ghnb + BT * Hsz;              // 32x256 h state master (f32)
  __bf16* hbf = (__bf16*)(hbuf + BT * Hsz);   // 32x256 h state bf16 mirror (GEMM A input)
  __bf16* xtb = hbf + BT * Hsz;               // 32x64 padded xt tile (bf16)

  const int tid = threadIdx.x, lane = tid & 31, wave = tid >> 5;
  const int b0 = blockIdx.x * BT;

  for (int i = tid; i < BT * Hsz; i += BLOCK) { hbuf[i] = 0.0f; hbf[i] = (__bf16)0.0f; }
  __syncthreads();

  for (int s = 0; s < Tsz; ++s) {
    const int to = Tsz - 1 - s;  // original time index (reversed scan)
    const float tdiff = (to > 0) ? (tglob[to] - tglob[to - 1]) : 0.0f;
    for (int i = tid; i < BT * 64; i += BLOCK) {
      int m = i >> 6, k = i & 63;
      float v = 0.0f;
      if (k < Dsz) v = x[((size_t)(b0 + m) * Tsz + to) * Dsz + k];
      else if (k == Dsz) v = tdiff;
      xtb[i] = (__bf16)v;
    }
    __syncthreads();

    // 2 x 48 output tiles over (32 x 768)
    for (int t = wave; t < 2 * 48; t += NWAVES) {
      int mi = (t / 48) << 4;
      int nc = (t % 48) << 4;     // global column 0..767
      int cg = nc >> 8;           // 0=r, 1=z, 2=n
      int nl = nc & 255;
      if (cg < 2) {
        // r/z gates: gi+gh fused in one accumulator, init with both biases
        float bv = b_ih[nc + (lane & 15)] + b_hh[nc + (lane & 15)];
        v8f c;
#pragma unroll
        for (int r = 0; r < 8; ++r) c[r] = bv;
        for (int k = 0; k < 64; k += 32) {
          v16bf a = load_a_bf16(xtb, 64, mi, k, lane);
          v16bf b = load_b_w(Wihp, 64, nc, k, lane);
          c = WMMA_BF16(a, b, c);
        }
        for (int k = 0; k < Hsz; k += 32) {
          v16bf a = load_a_bf16(hbf, Hsz, mi, k, lane);
          v16bf b = load_b_w(Whh, Hsz, nc, k, lane);
          c = WMMA_BF16(a, b, c);
        }
        store_c<3>(cg == 0 ? rbuf : zbuf, Hsz, mi, nl, lane, c);
      } else {
        // n gate needs gi_n and gh_n separately (n = tanh(i_n + r*h_n))
        v8f ci = bias_init(b_ih + 2 * Hsz, nl, lane);
        v8f ch = bias_init(b_hh + 2 * Hsz, nl, lane);
        for (int k = 0; k < 64; k += 32) {
          v16bf a = load_a_bf16(xtb, 64, mi, k, lane);
          v16bf b = load_b_w(Wihp, 64, nc, k, lane);
          ci = WMMA_BF16(a, b, ci);
        }
        for (int k = 0; k < Hsz; k += 32) {
          v16bf a = load_a_bf16(hbf, Hsz, mi, k, lane);
          v16bf b = load_b_w(Whh, Hsz, nc, k, lane);
          ch = WMMA_BF16(a, b, ch);
        }
        store_c<0>(ginb, Hsz, mi, nl, lane, ci);
        store_c<0>(ghnb, Hsz, mi, nl, lane, ch);
      }
    }
    __syncthreads();

    for (int i = tid; i < BT * Hsz; i += BLOCK) {
      float r = rbuf[i], zg = zbuf[i];
      float nn = tanhf(ginb[i] + r * ghnb[i]);
      float hn = (1.0f - zg) * nn + zg * hbuf[i];
      hbuf[i] = hn;
      hbf[i] = (__bf16)hn;   // convert exactly once per element per step
    }
    __syncthreads();
  }
  for (int i = tid; i < BT * Hsz; i += BLOCK)
    hT[(size_t)(b0 + (i >> 8)) * Hsz + (i & 255)] = hbuf[i];
}

// ---------------- latent head: z, z_mean, z_log_var; seed zs[:,0,:] ----------------
__global__ void k_latent(const float* __restrict__ hT, const float* __restrict__ W_hl,
                         const float* __restrict__ b_hl, const float* __restrict__ eps,
                         float* __restrict__ out_z, float* __restrict__ out_mean,
                         float* __restrict__ out_lv, float* __restrict__ zsbuf) {
  int i = blockIdx.x * blockDim.x + threadIdx.x;  // over B*L
  if (i >= Bsz * Lsz) return;
  int b = i / Lsz, l = i % Lsz;
  const float* h = hT + (size_t)b * Hsz;
  float m = b_hl[l], lv = b_hl[l + Lsz];
  for (int k = 0; k < Hsz; ++k) {
    float hv = h[k];
    m  += hv * W_hl[(size_t)l * Hsz + k];
    lv += hv * W_hl[(size_t)(l + Lsz) * Hsz + k];
  }
  float z = m + eps[i] * expf(0.5f * lv);
  out_mean[i] = m;
  out_lv[i] = lv;
  out_z[i] = z;
  zsbuf[((size_t)b * Tsz) * Lsz + l] = z;
}

// ---------------- ODE RK3 scan (persistent per batch-tile) ----------------
// src: bf16 GEMM input; dst: f32 k-buffer
__device__ __forceinline__ void eval_f(const __bf16* src, float* dst, __bf16* actA, __bf16* actB,
                                       const __bf16* W1, const float* b1,
                                       const __bf16* W2, const float* b2,
                                       const __bf16* W3, const float* b3,
                                       const __bf16* W4, const float* b4) {
  wmma_layer_b<1>(src,  W1, b1, actA, BT, Hsz, Lsz); __syncthreads();
  wmma_layer_b<1>(actA, W2, b2, actB, BT, Hsz, Hsz); __syncthreads();
  wmma_layer_b<1>(actB, W3, b3, actA, BT, Hsz, Hsz); __syncthreads();
  wmma_layer_f<0>(actA, W4, b4, dst,  BT, Lsz, Hsz); __syncthreads();
}

__global__ void __launch_bounds__(BLOCK)
k_ode(const float* __restrict__ tglob,
      const __bf16* __restrict__ W1, const float* __restrict__ b1,
      const __bf16* __restrict__ W2, const float* __restrict__ b2,
      const __bf16* __restrict__ W3, const float* __restrict__ b3,
      const __bf16* __restrict__ W4, const float* __restrict__ b4,
      float* __restrict__ zsbuf) {
  extern __shared__ char smem[];
  float*  y     = (float*)smem;          // 32x64 f32 master state
  float*  k1    = y  + BT * Lsz;
  float*  k2    = k1 + BT * Lsz;
  float*  k3    = k2 + BT * Lsz;
  __bf16* ybf   = (__bf16*)(k3 + BT * Lsz);  // bf16 mirror of y (GEMM input)
  __bf16* ytmpb = ybf   + BT * Lsz;          // bf16 stage input (only ever a GEMM input)
  __bf16* actA  = ytmpb + BT * Lsz;          // 32x256 bf16
  __bf16* actB  = actA  + BT * Hsz;          // 32x256 bf16

  const int tid = threadIdx.x;
  const int b0 = blockIdx.x * BT;

  for (int i = tid; i < BT * Lsz; i += BLOCK) {
    float v = zsbuf[((size_t)(b0 + (i >> 6)) * Tsz) * Lsz + (i & 63)];
    y[i] = v;
    ybf[i] = (__bf16)v;
  }
  __syncthreads();

  for (int iv = 0; iv < Tsz - 1; ++iv) {
    float dt = tglob[iv + 1] - tglob[iv];
    float hstep = 0.5f * dt;  // dt / N_SUB
    for (int sub = 0; sub < 2; ++sub) {
      eval_f(ybf, k1, actA, actB, W1, b1, W2, b2, W3, b3, W4, b4);
      for (int i = tid; i < BT * Lsz; i += BLOCK)
        ytmpb[i] = (__bf16)(y[i] + 0.5f * hstep * k1[i]);
      __syncthreads();
      eval_f(ytmpb, k2, actA, actB, W1, b1, W2, b2, W3, b3, W4, b4);
      for (int i = tid; i < BT * Lsz; i += BLOCK)
        ytmpb[i] = (__bf16)(y[i] + 0.75f * hstep * k2[i]);
      __syncthreads();
      eval_f(ytmpb, k3, actA, actB, W1, b1, W2, b2, W3, b3, W4, b4);
      for (int i = tid; i < BT * Lsz; i += BLOCK) {
        float v = y[i] + hstep * ((2.0f / 9.0f) * k1[i] + (1.0f / 3.0f) * k2[i] + (4.0f / 9.0f) * k3[i]);
        y[i] = v;
        ybf[i] = (__bf16)v;
      }
      __syncthreads();
    }
    for (int i = tid; i < BT * Lsz; i += BLOCK)
      zsbuf[((size_t)(b0 + (i >> 6)) * Tsz + iv + 1) * Lsz + (i & 63)] = y[i];
  }
}

// ---------------- decoder: x_p = relu(zs@Wl2h^T+b) @ Wh2o^T + b ----------------
__global__ void __launch_bounds__(BLOCK)
k_dec(const float* __restrict__ zsbuf,
      const __bf16* __restrict__ Wl2h, const float* __restrict__ b_l2h,
      const __bf16* __restrict__ Wh2o, const float* __restrict__ b_h2o,
      float* __restrict__ xp) {
  extern __shared__ char smem[];
  __bf16* zrow = (__bf16*)smem;        // 32x64 bf16
  __bf16* actA = zrow + BT * Lsz;      // 32x256 bf16
  const int tid = threadIdx.x;
  const int r0 = blockIdx.x * BT;      // global row = b*T + t
  for (int i = tid; i < BT * Lsz; i += BLOCK)
    zrow[i] = (__bf16)zsbuf[(size_t)(r0 + (i >> 6)) * Lsz + (i & 63)];
  __syncthreads();
  wmma_layer_b<2>(zrow, Wl2h, b_l2h, actA, BT, Hsz, Lsz);
  __syncthreads();
  wmma_layer_f<0>(actA, Wh2o, b_h2o, xp + (size_t)r0 * Dsz, BT, Dsz, Hsz);
}

// ---------------- host orchestration ----------------
extern "C" void kernel_launch(void* const* d_in, const int* in_sizes, int n_in,
                              void* d_out, int out_size, void* d_ws, size_t ws_size,
                              hipStream_t stream) {
  const float* x     = (const float*)d_in[0];
  const float* t     = (const float*)d_in[1];
  const float* eps   = (const float*)d_in[2];
  const float* W_ih  = (const float*)d_in[3];
  const float* b_ih  = (const float*)d_in[4];
  const float* W_hh  = (const float*)d_in[5];
  const float* b_hh  = (const float*)d_in[6];
  const float* W_hl  = (const float*)d_in[7];
  const float* b_hl  = (const float*)d_in[8];
  const float* W1    = (const float*)d_in[9];
  const float* b1    = (const float*)d_in[10];
  const float* W2    = (const float*)d_in[11];
  const float* b2    = (const float*)d_in[12];
  const float* W3    = (const float*)d_in[13];
  const float* b3    = (const float*)d_in[14];
  const float* W4    = (const float*)d_in[15];
  const float* b4    = (const float*)d_in[16];
  const float* W_l2h = (const float*)d_in[17];
  const float* b_l2h = (const float*)d_in[18];
  const float* W_h2o = (const float*)d_in[19];
  const float* b_h2o = (const float*)d_in[20];

  char* ws = (char*)d_ws;
  size_t off = 0;
  auto carve = [&](size_t bytes) -> void* {
    void* p = ws + off;
    off = (off + bytes + 255) & ~(size_t)255;
    return p;
  };
  __bf16* Wihp  = (__bf16*)carve((size_t)768 * 64 * 2);
  __bf16* Whhb  = (__bf16*)carve((size_t)768 * 256 * 2);
  __bf16* W1b   = (__bf16*)carve((size_t)256 * 64 * 2);
  __bf16* W2b   = (__bf16*)carve((size_t)256 * 256 * 2);
  __bf16* W3b   = (__bf16*)carve((size_t)256 * 256 * 2);
  __bf16* W4b   = (__bf16*)carve((size_t)64 * 256 * 2);
  __bf16* Wl2hb = (__bf16*)carve((size_t)256 * 64 * 2);
  __bf16* Wh2ob = (__bf16*)carve((size_t)32 * 256 * 2);
  float*  hT    = (float*)carve((size_t)Bsz * Hsz * 4);
  float*  zsb   = (float*)carve((size_t)Bsz * Tsz * Lsz * 4);

  float* out_xp   = (float*)d_out;                 // (B,T,D)
  float* out_z    = out_xp + (size_t)Bsz * Tsz * Dsz;
  float* out_mean = out_z + (size_t)Bsz * Lsz;
  float* out_lv   = out_mean + (size_t)Bsz * Lsz;

  // weight conversion (once per call; tiny)
  k_pad_wih<<<(768 * 64 + 255) / 256, 256, 0, stream>>>(W_ih, Wihp);
  k_cvt<<<(768 * 256 + 255) / 256, 256, 0, stream>>>(W_hh, Whhb, 768 * 256);
  k_cvt<<<(256 * 64 + 255) / 256, 256, 0, stream>>>(W1, W1b, 256 * 64);
  k_cvt<<<(256 * 256 + 255) / 256, 256, 0, stream>>>(W2, W2b, 256 * 256);
  k_cvt<<<(256 * 256 + 255) / 256, 256, 0, stream>>>(W3, W3b, 256 * 256);
  k_cvt<<<(64 * 256 + 255) / 256, 256, 0, stream>>>(W4, W4b, 64 * 256);
  k_cvt<<<(256 * 64 + 255) / 256, 256, 0, stream>>>(W_l2h, Wl2hb, 256 * 64);
  k_cvt<<<(32 * 256 + 255) / 256, 256, 0, stream>>>(W_h2o, Wh2ob, 32 * 256);

  // LDS: 5*32KB f32 + 16KB hbf + 4KB xtb = ~184 KB (< 320KB WGP LDS)
  size_t gruLds = (size_t)(5 * BT * Hsz) * 4 + (size_t)BT * Hsz * 2 + (size_t)BT * 64 * 2;
  k_gru<<<Bsz / BT, BLOCK, gruLds, stream>>>(x, t, Wihp, b_ih, Whhb, b_hh, hT);

  k_latent<<<(Bsz * Lsz + 255) / 256, 256, 0, stream>>>(hT, W_hl, b_hl, eps,
                                                        out_z, out_mean, out_lv, zsb);

  // LDS: 4*8KB f32 + 2*4KB bf16 + 2*16KB bf16 = 72 KB
  size_t odeLds = (size_t)(4 * BT * Lsz) * 4 + (size_t)(2 * BT * Lsz) * 2 + (size_t)(2 * BT * Hsz) * 2;
  k_ode<<<Bsz / BT, BLOCK, odeLds, stream>>>(t, W1b, b1, W2b, b2, W3b, b3, W4b, b4, zsb);

  // LDS: 4KB + 16KB bf16 = 20 KB
  size_t decLds = (size_t)BT * Lsz * 2 + (size_t)BT * Hsz * 2;
  k_dec<<<(Bsz * Tsz) / BT, BLOCK, decLds, stream>>>(zsb, Wl2hb, b_l2h, Wh2ob, b_h2o, out_xp);
}
